// FmriEncoder_MoE_77360950935854
// MI455X (gfx1250) — matrix operand order repaired and checked by reading, not compiled, not run on hardware
//
#include <hip/hip_runtime.h>

// MoE routed forward for MI455X (gfx1250, wave32, WMMA).
// B=4, N=512, H=3072, E=4, F=12288, O=4096. Top-1 routing => each token
// through exactly one expert. bf16 WMMA (f32 accum), fp32 weight stream.
// Block tile 128x128 (8 waves, each 32x64 = 8 accumulators), K-step 32.
// Grid: blockIdx.x = M-tile (fastest) so all M-tiles sweep one weight
// stripe back-to-back -> stripe read once from HBM, re-reads hit L2.

#define Bsz  4
#define Nseq 512
#define Hdim 3072
#define Edim 4
#define Fdim 12288
#define Odim 4096
#define TOK  (Bsz * Nseq)   // 2048

#define TM 128
#define TN 128
#define TK 32
#define LDP 40              // padded LDS row stride (elements), 80B = 16B-aligned

typedef __attribute__((ext_vector_type(16))) __bf16 v16bf;
typedef __attribute__((ext_vector_type(8)))  __bf16 v8bf;
typedef __attribute__((ext_vector_type(8)))  float  v8f;

__device__ __forceinline__ float gelu_exact(float x) {
    return 0.5f * x * (1.0f + erff(x * 0.70710678118654752440f));
}

// ---------------------------------------------------------------- router ----
__global__ __launch_bounds__(256) void router_kernel(
    const float* __restrict__ x, const float* __restrict__ Wr,
    const float* __restrict__ br, int* __restrict__ eid,
    float* __restrict__ gatep, int* __restrict__ counts)
{
    const int tok  = (blockIdx.x * blockDim.x + threadIdx.x) >> 5;
    const int lane = threadIdx.x & 31;
    if (tok >= TOK) return;

    const float* xr = x + (size_t)tok * Hdim;
    float a0 = 0.f, a1 = 0.f, a2 = 0.f, a3 = 0.f;
    for (int h = lane; h < Hdim; h += 32) {
        const float xv = xr[h];
        a0 += xv * Wr[0 * Hdim + h];
        a1 += xv * Wr[1 * Hdim + h];
        a2 += xv * Wr[2 * Hdim + h];
        a3 += xv * Wr[3 * Hdim + h];
    }
    #pragma unroll
    for (int off = 16; off > 0; off >>= 1) {
        a0 += __shfl_xor(a0, off, 32);
        a1 += __shfl_xor(a1, off, 32);
        a2 += __shfl_xor(a2, off, 32);
        a3 += __shfl_xor(a3, off, 32);
    }
    if (lane == 0) {
        float l[4] = { a0 + br[0], a1 + br[1], a2 + br[2], a3 + br[3] };
        int best = 0;
        #pragma unroll
        for (int e = 1; e < 4; ++e) if (l[e] > l[best]) best = e;  // first-max
        const float m = l[best];
        float s = 0.f;
        #pragma unroll
        for (int e = 0; e < 4; ++e) s += expf(l[e] - m);
        const float pt = 1.0f / s;          // softmax prob of the argmax
        eid[tok]   = best;
        gatep[tok] = pt / (pt + 1e-8f);
        atomicAdd(&counts[best], 1);
    }
}

__global__ void zero_counts_kernel(int* __restrict__ counts) {
    if (threadIdx.x < 8) counts[threadIdx.x] = 0;
}

// Deterministic compaction: offsets + per-expert ordered token lists.
__global__ void scan_fill_kernel(
    const int* __restrict__ counts, int* __restrict__ offsets,
    const int* __restrict__ eid, const float* __restrict__ gatep,
    int* __restrict__ perm, float* __restrict__ rowgate)
{
    if (threadIdx.x == 0 && blockIdx.x == 0) {
        int off = 0, cur[4];
        for (int e = 0; e < 4; ++e) { offsets[e] = off; cur[e] = off; off += counts[e]; }
        for (int t = 0; t < TOK; ++t) {
            const int e = eid[t];
            const int p = cur[e]++;
            perm[p]    = t;
            rowgate[p] = gatep[t];
        }
    }
}

// fp32 tokens -> bf16 rows in routed order (packed 8B stores).
__global__ __launch_bounds__(256) void gather_kernel(
    const float* __restrict__ x, const int* __restrict__ perm,
    __bf16* __restrict__ Xg)
{
    const int idx = (blockIdx.x * 256 + threadIdx.x) * 4;    // < TOK*Hdim
    const int pos = idx / Hdim;
    const int h   = idx - pos * Hdim;
    const int tok = perm[pos];
    const float4 v = *reinterpret_cast<const float4*>(x + (size_t)tok * Hdim + h);
    union { __bf16 h4[4]; uint2 u; } p;
    p.h4[0] = (__bf16)v.x; p.h4[1] = (__bf16)v.y;
    p.h4[2] = (__bf16)v.z; p.h4[3] = (__bf16)v.w;
    *reinterpret_cast<uint2*>(Xg + (size_t)pos * Hdim + h) = p.u;
}

// ------------------------------------------------------------ WMMA GEMMs ----
// Fragment loads per ISA 7.12.2 (wave32).
__device__ __forceinline__ v16bf load_afrag(const __bf16* __restrict__ row,
                                            int half) {
    v16bf a;
    const int kb1 = half * 8, kb2 = 16 + half * 8;
    #pragma unroll
    for (int j = 0; j < 8; ++j) a[j] = row[kb1 + j];
    #pragma unroll
    for (int j = 0; j < 8; ++j) a[8 + j] = row[kb2 + j];
    return a;
}
__device__ __forceinline__ v16bf load_bfrag(const __bf16* __restrict__ col,
                                            int half) {
    v16bf b;
    const int kb1 = half * 8, kb2 = 16 + half * 8;
    #pragma unroll
    for (int j = 0; j < 8; ++j) b[j] = col[kb1 + j];
    #pragma unroll
    for (int j = 0; j < 8; ++j) b[8 + j] = col[kb2 + j];
    return b;
}

// GEMM1: Hmid[rows,F] = GELU(Xg[rows,H] @ W1[e] + b1[e]), bf16 out.
__global__ __launch_bounds__(256) void gemm1_gelu_kernel(
    const __bf16* __restrict__ Xg, const float* __restrict__ W1,
    const float* __restrict__ b1, const int* __restrict__ counts,
    const int* __restrict__ offsets, __bf16* __restrict__ Hmid)
{
    __shared__ __bf16 sA[TM * LDP];
    __shared__ __bf16 sB[TN * LDP];

    const int e      = blockIdx.z;
    const int cnt    = counts[e];
    const int m_base = blockIdx.x * TM;            // M fastest-varying
    if (m_base >= cnt) return;
    const int row0   = offsets[e] + m_base;
    const int n_base = blockIdx.y * TN;

    const int tid  = threadIdx.x;
    const int lane = tid & 31;
    const int wave = tid >> 5;
    const int wm   = wave & 3;                     // 4 M-groups of 32
    const int wn   = wave >> 2;                    // 2 N-groups of 64
    const int ml   = lane & 15;
    const int half = lane >> 4;

    v8f acc[2][4];
    #pragma unroll
    for (int i = 0; i < 2; ++i)
        #pragma unroll
        for (int j = 0; j < 4; ++j)
            #pragma unroll
            for (int r = 0; r < 8; ++r) acc[i][j][r] = 0.0f;

    const int am  = tid >> 1;                      // A: 2 threads/row, 16 bf16 each
    const int ak  = (tid & 1) << 4;
    const int bkp = (tid & 15) << 1;               // B: k-pair (even k)
    const int bn  = (tid >> 4) << 3;               // 8 n-values per thread
    const bool arow_ok = (m_base + am < cnt);
    const __bf16* aptr = Xg + (size_t)(row0 + am) * Hdim + ak;
    const size_t wbase = (size_t)e * Hdim * Fdim;

    for (int kk = 0; kk < Hdim; kk += TK) {
        // ---- stage A (rows beyond count -> zero) ----
        v8bf av0, av1;
        if (arow_ok) {
            av0 = *reinterpret_cast<const v8bf*>(aptr + kk);
            av1 = *reinterpret_cast<const v8bf*>(aptr + kk + 8);
        } else {
            #pragma unroll
            for (int j = 0; j < 8; ++j) { av0[j] = (__bf16)0.0f; av1[j] = (__bf16)0.0f; }
        }
        *reinterpret_cast<v8bf*>(&sA[am * LDP + ak])     = av0;
        *reinterpret_cast<v8bf*>(&sA[am * LDP + ak + 8]) = av1;

        // ---- stage B transposed: sB[n][k], packed bf16 pairs ----
        const float* brow0 = W1 + wbase + (size_t)(kk + bkp) * Fdim + n_base + bn;
        const float* brow1 = brow0 + Fdim;
        float v0[8], v1[8];
        #pragma unroll
        for (int j = 0; j < 8; ++j) { v0[j] = brow0[j]; v1[j] = brow1[j]; }
        #pragma unroll
        for (int j = 0; j < 8; ++j) {
            union { __bf16 h2[2]; unsigned u; } p;
            p.h2[0] = (__bf16)v0[j];
            p.h2[1] = (__bf16)v1[j];
            *reinterpret_cast<unsigned*>(&sB[(bn + j) * LDP + bkp]) = p.u;
        }
        if (kk + TK < Hdim)
            __builtin_prefetch(brow0 + (size_t)TK * Fdim, 0, 0);
        __syncthreads();

        v16bf afr[2], bfr[4];
        #pragma unroll
        for (int mi = 0; mi < 2; ++mi)
            afr[mi] = load_afrag(&sA[(wm * 32 + mi * 16 + ml) * LDP], half);
        #pragma unroll
        for (int ni = 0; ni < 4; ++ni)
            bfr[ni] = load_bfrag(&sB[(wn * 64 + ni * 16 + ml) * LDP], half);
        #pragma unroll
        for (int mi = 0; mi < 2; ++mi)
            #pragma unroll
            for (int ni = 0; ni < 4; ++ni)
                acc[mi][ni] = __builtin_amdgcn_wmma_f32_16x16x32_bf16(
                    false, afr[mi], false, bfr[ni], (short)0, acc[mi][ni],
                    false, false);
        __syncthreads();
    }

    float bias[4];
    #pragma unroll
    for (int ni = 0; ni < 4; ++ni)
        bias[ni] = b1[e * Fdim + n_base + wn * 64 + ni * 16 + ml];
    #pragma unroll
    for (int mi = 0; mi < 2; ++mi)
        #pragma unroll
        for (int r = 0; r < 8; ++r) {
            const int mrow = wm * 32 + mi * 16 + half * 8 + r;
            if (m_base + mrow < cnt) {
                __bf16* dst = Hmid + (size_t)(row0 + mrow) * Fdim
                                   + n_base + wn * 64 + ml;
                #pragma unroll
                for (int ni = 0; ni < 4; ++ni)
                    dst[ni * 16] = (__bf16)gelu_exact(acc[mi][ni][r] + bias[ni]);
            }
        }
}

// GEMM2: out[b, o, n] = gate * (Hmid[row,:] @ W2[e] + b2[e])  (scatter, [B,O,N])
__global__ __launch_bounds__(256) void gemm2_scatter_kernel(
    const __bf16* __restrict__ Hmid, const float* __restrict__ W2,
    const float* __restrict__ b2, const int* __restrict__ counts,
    const int* __restrict__ offsets, const int* __restrict__ perm,
    const float* __restrict__ rowgate, float* __restrict__ out)
{
    __shared__ __bf16 sA[TM * LDP];
    __shared__ __bf16 sB[TN * LDP];

    const int e      = blockIdx.z;
    const int cnt    = counts[e];
    const int m_base = blockIdx.x * TM;            // M fastest-varying
    if (m_base >= cnt) return;
    const int row0   = offsets[e] + m_base;
    const int n_base = blockIdx.y * TN;

    const int tid  = threadIdx.x;
    const int lane = tid & 31;
    const int wave = tid >> 5;
    const int wm   = wave & 3;
    const int wn   = wave >> 2;
    const int ml   = lane & 15;
    const int half = lane >> 4;

    v8f acc[2][4];
    #pragma unroll
    for (int i = 0; i < 2; ++i)
        #pragma unroll
        for (int j = 0; j < 4; ++j)
            #pragma unroll
            for (int r = 0; r < 8; ++r) acc[i][j][r] = 0.0f;

    const int am  = tid >> 1;
    const int ak  = (tid & 1) << 4;
    const int bkp = (tid & 15) << 1;
    const int bn  = (tid >> 4) << 3;
    const bool arow_ok = (m_base + am < cnt);
    const __bf16* aptr = Hmid + (size_t)(row0 + am) * Fdim + ak;
    const size_t wbase = (size_t)e * Fdim * Odim;

    for (int kk = 0; kk < Fdim; kk += TK) {
        v8bf av0, av1;
        if (arow_ok) {
            av0 = *reinterpret_cast<const v8bf*>(aptr + kk);
            av1 = *reinterpret_cast<const v8bf*>(aptr + kk + 8);
        } else {
            #pragma unroll
            for (int j = 0; j < 8; ++j) { av0[j] = (__bf16)0.0f; av1[j] = (__bf16)0.0f; }
        }
        *reinterpret_cast<v8bf*>(&sA[am * LDP + ak])     = av0;
        *reinterpret_cast<v8bf*>(&sA[am * LDP + ak + 8]) = av1;

        const float* brow0 = W2 + wbase + (size_t)(kk + bkp) * Odim + n_base + bn;
        const float* brow1 = brow0 + Odim;
        float v0[8], v1[8];
        #pragma unroll
        for (int j = 0; j < 8; ++j) { v0[j] = brow0[j]; v1[j] = brow1[j]; }
        #pragma unroll
        for (int j = 0; j < 8; ++j) {
            union { __bf16 h2[2]; unsigned u; } p;
            p.h2[0] = (__bf16)v0[j];
            p.h2[1] = (__bf16)v1[j];
            *reinterpret_cast<unsigned*>(&sB[(bn + j) * LDP + bkp]) = p.u;
        }
        if (kk + TK < Fdim)
            __builtin_prefetch(brow0 + (size_t)TK * Odim, 0, 0);
        __syncthreads();

        v16bf afr[2], bfr[4];
        #pragma unroll
        for (int mi = 0; mi < 2; ++mi)
            afr[mi] = load_afrag(&sA[(wm * 32 + mi * 16 + ml) * LDP], half);
        #pragma unroll
        for (int ni = 0; ni < 4; ++ni)
            bfr[ni] = load_bfrag(&sB[(wn * 64 + ni * 16 + ml) * LDP], half);
        #pragma unroll
        for (int mi = 0; mi < 2; ++mi)
            #pragma unroll
            for (int ni = 0; ni < 4; ++ni)
                acc[mi][ni] = __builtin_amdgcn_wmma_f32_16x16x32_bf16(
                    false, afr[mi], false, bfr[ni], (short)0, acc[mi][ni],
                    false, false);
        __syncthreads();
    }

    float bias[4];
    #pragma unroll
    for (int ni = 0; ni < 4; ++ni)
        bias[ni] = b2[e * Odim + n_base + wn * 64 + ni * 16 + ml];
    #pragma unroll
    for (int mi = 0; mi < 2; ++mi)
        #pragma unroll
        for (int r = 0; r < 8; ++r) {
            const int mrow = wm * 32 + mi * 16 + half * 8 + r;
            if (m_base + mrow < cnt) {
                const int grow  = row0 + mrow;
                const int token = perm[grow];
                const float g   = rowgate[grow];
                float* dst = out + (size_t)(token >> 9) * ((size_t)Odim * Nseq)
                                 + (size_t)(n_base + wn * 64 + ml) * Nseq
                                 + (token & (Nseq - 1));
                #pragma unroll
                for (int ni = 0; ni < 4; ++ni)
                    dst[(size_t)(ni * 16) * Nseq] = (acc[mi][ni][r] + bias[ni]) * g;
            }
        }
}

// ---------------------------------------------------------------- launch ----
extern "C" void kernel_launch(void* const* d_in, const int* in_sizes, int n_in,
                              void* d_out, int out_size, void* d_ws, size_t ws_size,
                              hipStream_t stream) {
    const float* x  = (const float*)d_in[0];   // [B,N,H]
    const float* Wr = (const float*)d_in[1];   // [E,H]
    const float* br = (const float*)d_in[2];   // [E]
    const float* W1 = (const float*)d_in[3];   // [E,H,F]
    const float* b1 = (const float*)d_in[4];   // [E,F]
    const float* W2 = (const float*)d_in[5];   // [E,F,O]
    const float* b2 = (const float*)d_in[6];   // [E,O]
    float* out = (float*)d_out;                // [B,O,N]

    char* ws = (char*)d_ws;
    int*    eid     = (int*)ws;                 ws += (size_t)TOK * 4;
    float*  gatep   = (float*)ws;               ws += (size_t)TOK * 4;
    int*    counts  = (int*)ws;                 ws += 32;
    int*    offsets = (int*)ws;                 ws += 32;
    int*    perm    = (int*)ws;                 ws += (size_t)TOK * 4;
    float*  rowgate = (float*)ws;               ws += (size_t)TOK * 4;
    __bf16* Xg      = (__bf16*)ws;              ws += (size_t)TOK * Hdim * 2;
    __bf16* Hmid    = (__bf16*)ws;

    zero_counts_kernel<<<1, 32, 0, stream>>>(counts);
    router_kernel<<<TOK / 8, 256, 0, stream>>>(x, Wr, br, eid, gatep, counts);
    scan_fill_kernel<<<1, 1, 0, stream>>>(counts, offsets, eid, gatep, perm, rowgate);
    gather_kernel<<<(TOK * Hdim / 4) / 256, 256, 0, stream>>>(x, perm, Xg);

    dim3 g1(TOK / TM, Fdim / TN, Edim);   // 16 x 96 x 4  (M fastest)
    gemm1_gelu_kernel<<<g1, 256, 0, stream>>>(Xg, W1, b1, counts, offsets, Hmid);

    dim3 g2(TOK / TM, Odim / TN, Edim);   // 16 x 32 x 4  (M fastest)
    gemm2_scatter_kernel<<<g2, 256, 0, stream>>>(Hmid, W2, b2, counts, offsets,
                                                 perm, rowgate, out);
}
